// mod_RGCN_24661702214220
// MI455X (gfx1250) — matrix operand order repaired and checked by reading, not compiled, not run on hardware
//
#include <hip/hip_runtime.h>

#define N_USER 50000
#define N_FOOD 50000
#define N_ING  50000
#define N_CAT  5000
#define N_HAB  5000
#define N_NODES 160000
#define NEDGE  600000
#define NREL   8
#define D      128

typedef __bf16 bf16_t;
typedef __attribute__((ext_vector_type(16))) __bf16 v16bf;
typedef __attribute__((ext_vector_type(8)))  __bf16 v8bf;
typedef __attribute__((ext_vector_type(4)))  __bf16 v4bf;
typedef __attribute__((ext_vector_type(8)))  float  v8f;
typedef __attribute__((ext_vector_type(4)))  float  v4f;

// ---------------------------------------------------------------------------
// Weight prep: fp32 row-major (K x 128) -> bf16 column-major WT[c*K + k],
// nmat independent matrices back to back (for the (R,128,128) stacks).
// ---------------------------------------------------------------------------
__global__ void transpose_w_bf16_kernel(const float* __restrict__ W,
                                        bf16_t* __restrict__ WT,
                                        int K, int nmat)
{
    int i = blockIdx.x * blockDim.x + threadIdx.x;
    int total = nmat * K * D;
    if (i >= total) return;
    int m   = i / (K * D);
    int rem = i - m * K * D;
    int k = rem / D;
    int c = rem - k * D;
    WT[(size_t)m * K * D + (size_t)c * K + k] = (bf16_t)W[i];
}

// ---------------------------------------------------------------------------
// GEMM: out[out_row_off + m, 0:128] = A[m, 0:K] @ W[0:K, 0:128] (+ bias)
// A row-major (lda=K), fp32 or bf16.  WT bf16 COLUMN-major (c*K + k).
// Block = 256 threads (8 waves); block tile 64 rows x 128 cols.
// Wave (w>>1) -> 16-row band, (w&1) -> 64-col half -> 4 WMMA tiles per wave,
// all sharing one A fragment.  K is compile-time: fully unrolled, all loads
// are base+imm-offset b128s; fragments load before the WMMA chain so the
// scheduler can overlap loads with matrix ops.
// ---------------------------------------------------------------------------
template<int K, int A_BF16, int OUT_BF16, int HAS_BIAS>
__global__ __launch_bounds__(256)
void gemm_wmma_kernel(const void* __restrict__ A_, int M,
                      const bf16_t* __restrict__ WT,
                      const float* __restrict__ bias,
                      void* __restrict__ out_, int out_row_off)
{
    const int tid   = threadIdx.x;
    const int lane  = tid & 31;
    const int wave  = tid >> 5;
    const int row0  = blockIdx.x * 64;
    const int rbase = (wave >> 1) * 16;     // 0,16,32,48
    const int cbase = (wave & 1) * 64;      // 0 or 64
    const int kHalf = lane >> 4;            // 0: lanes 0-15, 1: lanes 16-31
    const int l15   = lane & 15;

    // A fragment row for this lane (ISA 16-bit A layout: lane%16 = M row,
    // lanes 0-15 carry K 0-7/16-23, lanes 16-31 carry K 8-15/24-31)
    const int  arow  = row0 + rbase + l15;
    const bool rowok = (arow < M);
    const int  arowc = rowok ? arow : 0;

    const float*  Af = (const float*)A_  + (size_t)arowc * K + kHalf * 8;
    const bf16_t* Ab = (const bf16_t*)A_ + (size_t)arowc * K + kHalf * 8;
    // B base: lane%16 = column, lanes 0-15 K 0-7/16-23, lanes 16-31 K 8-15/24-31.
    const bf16_t* Wb = WT + (size_t)(cbase + l15) * K + kHalf * 8;

    v8f acc[4] = {{}, {}, {}, {}};

    #pragma unroll
    for (int k0 = 0; k0 < K; k0 += 32) {
        // ---- load phase: A fragment + 4 B fragments into distinct regs ----
        v16bf a;
        {
            v8bf lo, hi;
            if (A_BF16) {
                lo = *(const v8bf*)(Ab + k0);          // K kb..kb+7   (16B aligned)
                hi = *(const v8bf*)(Ab + k0 + 16);     // K kb+16..kb+23
            } else {
                v8f f0 = *(const v8f*)(Af + k0);
                v8f f1 = *(const v8f*)(Af + k0 + 16);
                #pragma unroll
                for (int j = 0; j < 8; ++j) { lo[j] = (bf16_t)f0[j]; hi[j] = (bf16_t)f1[j]; }
            }
            a = __builtin_shufflevector(lo, hi, 0,1,2,3,4,5,6,7,8,9,10,11,12,13,14,15);
            if (!rowok) { v16bf z = {}; a = z; }
        }
        v16bf b[4];
        #pragma unroll
        for (int t = 0; t < 4; ++t) {
            const bf16_t* wt = Wb + k0 + (size_t)(16 * t) * K;
            v8bf b0 = *(const v8bf*)(wt);
            v8bf b1 = *(const v8bf*)(wt + 16);
            b[t] = __builtin_shufflevector(b0, b1, 0,1,2,3,4,5,6,7,8,9,10,11,12,13,14,15);
        }
        // ---- compute phase ----
        #pragma unroll
        for (int t = 0; t < 4; ++t)
            acc[t] = __builtin_amdgcn_wmma_f32_16x16x32_bf16(false, a, false, b[t],
                                                             (short)0, acc[t], false, false);
    }

    // C layout: VGPR i, lanes 0-15 -> (M=i, N=col), lanes 16-31 -> (M=8+i, N=col)
    #pragma unroll
    for (int t = 0; t < 4; ++t) {
        const int col = cbase + 16 * t + l15;
        const float bc = HAS_BIAS ? bias[col] : 0.0f;
        #pragma unroll
        for (int i = 0; i < 8; ++i) {
            int gr = row0 + rbase + kHalf * 8 + i;
            if (gr < M) {
                size_t o = (size_t)(out_row_off + gr) * D + col;
                float v = acc[t][i] + bc;
                if (OUT_BF16) ((bf16_t*)out_)[o] = (bf16_t)v;
                else          ((float*)out_)[o]  = v;
            }
        }
    }
}

// ---------------------------------------------------------------------------
// Edge-degree counting / inverse, scatter-mean, relu
// ---------------------------------------------------------------------------
__global__ void zero_i32_kernel(int* __restrict__ p, int n)
{
    int i = blockIdx.x * blockDim.x + threadIdx.x;
    if (i < n) p[i] = 0;
}

__global__ void count_edges_kernel(const int* __restrict__ dst,
                                   const int* __restrict__ et,
                                   int* __restrict__ cnt)
{
    int e = blockIdx.x * blockDim.x + threadIdx.x;
    if (e < NEDGE) atomicAdd(&cnt[et[e] * N_NODES + dst[e]], 1);
}

__global__ void inv_cnt_kernel(const int* __restrict__ cnt, float* __restrict__ inv)
{
    int i = blockIdx.x * blockDim.x + threadIdx.x;
    if (i < NREL * N_NODES) {
        int c = cnt[i];
        inv[i] = 1.0f / (float)(c > 1 ? c : 1);
    }
}

// one wave per edge; 4 bf16 per lane via one 8-byte load; mean folded via inv
__global__ __launch_bounds__(256)
void scatter_kernel(const int* __restrict__ src, const int* __restrict__ dst,
                    const int* __restrict__ et, int r,
                    const bf16_t* __restrict__ H, const float* __restrict__ inv,
                    float* __restrict__ out)
{
    int e    = (int)((blockIdx.x * blockDim.x + threadIdx.x) >> 5);
    int lane = threadIdx.x & 31;
    if (e >= NEDGE) return;
    if (et[e] != r) return;
    int s = src[e], d = dst[e];
    float sc = inv[r * N_NODES + d];
    v4bf hv = *(const v4bf*)(H + (size_t)s * D + lane * 4);
    float* o = out + (size_t)d * D + lane * 4;
    #pragma unroll
    for (int j = 0; j < 4; ++j)
        atomicAdd(&o[j], (float)hv[j] * sc);
}

__global__ void relu_to_bf16_kernel(const float* __restrict__ x, bf16_t* __restrict__ y, int n4)
{
    int i = blockIdx.x * blockDim.x + threadIdx.x;
    if (i < n4) {
        v4f v = *(const v4f*)(x + (size_t)i * 4);
        v4bf o;
        #pragma unroll
        for (int j = 0; j < 4; ++j) o[j] = (bf16_t)(v[j] > 0.0f ? v[j] : 0.0f);
        *(v4bf*)(y + (size_t)i * 4) = o;
    }
}

// ---------------------------------------------------------------------------
// Launcher
// ---------------------------------------------------------------------------
extern "C" void kernel_launch(void* const* d_in, const int* in_sizes, int n_in,
                              void* d_out, int out_size, void* d_ws, size_t ws_size,
                              hipStream_t stream)
{
    const float* x_user = (const float*)d_in[0];
    const float* x_food = (const float*)d_in[1];
    const float* x_ing  = (const float*)d_in[2];
    const float* x_cat  = (const float*)d_in[3];
    const float* x_hab  = (const float*)d_in[4];
    const int*   eidx   = (const int*)d_in[5];
    const int*   etyp   = (const int*)d_in[6];
    const float* Wu = (const float*)d_in[7];  const float* bu  = (const float*)d_in[8];
    const float* Wf = (const float*)d_in[9];  const float* bfo = (const float*)d_in[10];
    const float* Wi = (const float*)d_in[11]; const float* bi  = (const float*)d_in[12];
    const float* Wc = (const float*)d_in[13]; const float* bc  = (const float*)d_in[14];
    const float* Wh = (const float*)d_in[15]; const float* bh  = (const float*)d_in[16];
    const float* W1 = (const float*)d_in[17]; const float* root1 = (const float*)d_in[18];
    const float* bias1 = (const float*)d_in[19];
    const float* W2 = (const float*)d_in[20]; const float* root2 = (const float*)d_in[21];
    const float* bias2 = (const float*)d_in[22];

    const int* esrc = eidx;
    const int* edst = eidx + NEDGE;

    // ---- workspace layout ----
    char* ws = (char*)d_ws;
    size_t off = 0;
    bf16_t* XB = (bf16_t*)(ws + off); off += (size_t)N_NODES * D * sizeof(bf16_t);
    off = (off + 255) & ~(size_t)255;
    bf16_t* H  = (bf16_t*)(ws + off); off += (size_t)N_NODES * D * sizeof(bf16_t);
    off = (off + 255) & ~(size_t)255;
    int*    cnt = (int*)(ws + off);   off += (size_t)NREL * N_NODES * sizeof(int);
    off = (off + 255) & ~(size_t)255;
    float*  inv = (float*)(ws + off); off += (size_t)NREL * N_NODES * sizeof(float);
    off = (off + 255) & ~(size_t)255;
    bf16_t* WB = (bf16_t*)(ws + off);
    size_t wo = 0;
    bf16_t* WuT = WB + wo; wo += (size_t)256 * D;
    bf16_t* WfT = WB + wo; wo += (size_t)128 * D;
    bf16_t* WiT = WB + wo; wo += (size_t)128 * D;
    bf16_t* WcT = WB + wo; wo += (size_t)64  * D;
    bf16_t* WhT = WB + wo; wo += (size_t)64  * D;
    bf16_t* R1T = WB + wo; wo += (size_t)128 * D;
    bf16_t* R2T = WB + wo; wo += (size_t)128 * D;
    bf16_t* W1T = WB + wo; wo += (size_t)NREL * 128 * D;
    bf16_t* W2T = WB + wo; wo += (size_t)NREL * 128 * D;

    float* Y = (float*)d_out;

    dim3 blk(256);
    auto cdiv = [](int a, int b) { return (a + b - 1) / b; };

    // ---- weight prep: fp32 row-major -> bf16 column-major (done once/launch) ----
    transpose_w_bf16_kernel<<<cdiv(256*D, 256),      blk, 0, stream>>>(Wu,    WuT, 256, 1);
    transpose_w_bf16_kernel<<<cdiv(128*D, 256),      blk, 0, stream>>>(Wf,    WfT, 128, 1);
    transpose_w_bf16_kernel<<<cdiv(128*D, 256),      blk, 0, stream>>>(Wi,    WiT, 128, 1);
    transpose_w_bf16_kernel<<<cdiv(64*D, 256),       blk, 0, stream>>>(Wc,    WcT, 64,  1);
    transpose_w_bf16_kernel<<<cdiv(64*D, 256),       blk, 0, stream>>>(Wh,    WhT, 64,  1);
    transpose_w_bf16_kernel<<<cdiv(128*D, 256),      blk, 0, stream>>>(root1, R1T, 128, 1);
    transpose_w_bf16_kernel<<<cdiv(128*D, 256),      blk, 0, stream>>>(root2, R2T, 128, 1);
    transpose_w_bf16_kernel<<<cdiv(NREL*128*D, 256), blk, 0, stream>>>(W1,    W1T, 128, NREL);
    transpose_w_bf16_kernel<<<cdiv(NREL*128*D, 256), blk, 0, stream>>>(W2,    W2T, 128, NREL);

    // ---- input projections -> XB (bf16), bias fused ----
    gemm_wmma_kernel<256,0,1,1><<<cdiv(N_USER,64), blk, 0, stream>>>(x_user, N_USER, WuT, bu,  XB, 0);
    gemm_wmma_kernel<128,0,1,1><<<cdiv(N_FOOD,64), blk, 0, stream>>>(x_food, N_FOOD, WfT, bfo, XB, N_USER);
    gemm_wmma_kernel<128,0,1,1><<<cdiv(N_ING, 64), blk, 0, stream>>>(x_ing,  N_ING,  WiT, bi,  XB, N_USER + N_FOOD);
    gemm_wmma_kernel<64, 0,1,1><<<cdiv(N_CAT, 64), blk, 0, stream>>>(x_cat,  N_CAT,  WcT, bc,  XB, N_USER + N_FOOD + N_ING);
    gemm_wmma_kernel<64, 0,1,1><<<cdiv(N_HAB, 64), blk, 0, stream>>>(x_hab,  N_HAB,  WhT, bh,  XB, N_USER + N_FOOD + N_ING + N_CAT);

    // ---- per-(relation,dst) in-degree -> inverse (shared by both layers) ----
    const int gRN = cdiv(NREL * N_NODES, 256);
    zero_i32_kernel   <<<gRN, blk, 0, stream>>>(cnt, NREL * N_NODES);
    count_edges_kernel<<<cdiv(NEDGE, 256), blk, 0, stream>>>(edst, etyp, cnt);
    inv_cnt_kernel    <<<gRN, blk, 0, stream>>>(cnt, inv);

    const int gN = cdiv(N_NODES, 64);   // 2500 blocks for the big GEMMs
    const int gE = cdiv(NEDGE, 8);      // 1 wave/edge, 8 waves/block

    // ---- layer 1: Y = XB@root1 + bias1 + sum_r mean_r(XB@W1_r) ----
    gemm_wmma_kernel<128,1,0,1><<<gN, blk, 0, stream>>>(XB, N_NODES, R1T, bias1, Y, 0);
    for (int r = 0; r < NREL; ++r) {
        gemm_wmma_kernel<128,1,1,0><<<gN, blk, 0, stream>>>(XB, N_NODES, W1T + (size_t)r * 128 * D, nullptr, H, 0);
        scatter_kernel<<<gE, blk, 0, stream>>>(esrc, edst, etyp, r, H, inv, Y);
    }

    // ---- relu -> XB (bf16) ----
    relu_to_bf16_kernel<<<cdiv(N_NODES * D / 4, 256), blk, 0, stream>>>(Y, XB, N_NODES * D / 4);

    // ---- layer 2 (reuses d_out as accumulator; reads only XB) ----
    gemm_wmma_kernel<128,1,0,1><<<gN, blk, 0, stream>>>(XB, N_NODES, R2T, bias2, Y, 0);
    for (int r = 0; r < NREL; ++r) {
        gemm_wmma_kernel<128,1,1,0><<<gN, blk, 0, stream>>>(XB, N_NODES, W2T + (size_t)r * 128 * D, nullptr, H, 0);
        scatter_kernel<<<gE, blk, 0, stream>>>(esrc, edst, etyp, r, H, inv, Y);
    }
}
